// ParallelMoELinear_83081847374408
// MI455X (gfx1250) — matrix operand order repaired and checked by reading, not compiled, not run on hardware
//
#include <hip/hip_runtime.h>
#include <stdint.h>

// CDNA5 (gfx1250) wave32 WMMA types
typedef __attribute__((ext_vector_type(16))) __bf16 v16bf;
typedef __attribute__((ext_vector_type(8)))  float  v8f;

#define T_TOKENS 8192
#define IN_F     1024
#define OUT_F    1024
#define N_EXP    8

#define TILE_M 128
#define TILE_N 128
#define TILE_K 32
#define KSTEPS (IN_F / TILE_K)

union FragU { uint4 q[2]; v16bf v; };

__device__ __forceinline__ uint32_t pack_bf16x2(__bf16 a, __bf16 b) {
  union { __bf16 h[2]; uint32_t u; } u;
  u.h[0] = a; u.h[1] = b;
  return u.u;
}

// Split f32 into leading bf16 (RNE) + bf16 residual: x ~= h + l (~16-bit mantissa).
__device__ __forceinline__ void split_hilo(float x, __bf16 &h, __bf16 &l) {
  h = (__bf16)x;
  l = (__bf16)(x - (float)h);
}

// Minimal workgroup split-barrier: waits only DScnt (NOT LOADcnt), so in-flight
// global register-prefetch loads keep flying across the barrier.
__device__ __forceinline__ void wg_barrier_ds() {
  asm volatile("s_wait_dscnt 0x0\n\t"
               "s_barrier_signal -1\n\t"
               "s_barrier_wait -1" ::: "memory");
}

__global__ __launch_bounds__(256)
void moe_grouped_gemm_bf16x3(const float* __restrict__ X,   // [T, IN_F]
                             const int*   __restrict__ gsz, // [N_EXP]
                             const float* __restrict__ W,   // [N_EXP, IN_F, OUT_F]
                             const float* __restrict__ Bias,// [N_EXP, OUT_F]
                             float*       __restrict__ Y)   // [T, OUT_F]
{
  // Double-buffered LDS staging: A row-major [m][k], B transposed [n][k];
  // bf16 packed 2/dword. 4 arrays x 16 KB = 64 KB (<< 320 KB/WGP).
  __shared__ __align__(16) uint32_t AhiU[2][TILE_M][TILE_K / 2];
  __shared__ __align__(16) uint32_t AloU[2][TILE_M][TILE_K / 2];
  __shared__ __align__(16) uint32_t BhiU[2][TILE_N][TILE_K / 2];
  __shared__ __align__(16) uint32_t BloU[2][TILE_N][TILE_K / 2];

  const int m0   = blockIdx.y * TILE_M;
  const int n0   = blockIdx.x * TILE_N;
  const int tid  = threadIdx.x;
  const int lane = tid & 31;
  const int wave = tid >> 5;
  const int wm   = wave >> 1;   // 0..3 : 32-row slab within tile
  const int wn   = wave & 1;    // 0..1 : 64-col slab within tile

  // Per-thread staging coordinates (constant across K-steps).
  const int a_row[4] = { (tid + 0*256) >> 3, (tid + 1*256) >> 3,
                         (tid + 2*256) >> 3, (tid + 3*256) >> 3 };
  const int a_kc     = (tid & 7) << 2;                 // same for all i
  const int b_kk[2]  = { (tid >> 5) << 1, ((tid + 256) >> 5) << 1 };
  const int b_nn     = (tid & 31) << 2;                // same for both i

  // Ragged group prefix offsets (8 scalar loads, uniform across block).
  int offs[N_EXP + 1];
  offs[0] = 0;
#pragma unroll
  for (int e = 0; e < N_EXP; ++e) offs[e + 1] = offs[e] + gsz[e];

  for (int e = 0; e < N_EXP; ++e) {
    const int segLo = offs[e]     > m0          ? offs[e]     : m0;
    const int segHi = offs[e + 1] < m0 + TILE_M ? offs[e + 1] : m0 + TILE_M;
    if (segLo >= segHi) continue;   // uniform branch: whole block skips together

    const float* We = W + ((size_t)e << 20);   // e * IN_F * OUT_F

    // ---- prologue: global register-prefetch of K-tile 0 ----
    float4 ra[4], rb0[2], rb1[2];
#pragma unroll
    for (int i = 0; i < 4; ++i)
      ra[i] = *(const float4*)(X + (size_t)(m0 + a_row[i]) * IN_F + a_kc);
#pragma unroll
    for (int i = 0; i < 2; ++i) {
      const float* p0 = We + (size_t)b_kk[i] * OUT_F + n0 + b_nn;
      rb0[i] = *(const float4*)(p0);
      rb1[i] = *(const float4*)(p0 + OUT_F);
    }

    v8f acc[2][4] = {};

#pragma unroll 2
    for (int ks = 0; ks < KSTEPS; ++ks) {
      const int buf = ks & 1;

      // ---- convert staged regs -> bf16 hi/lo, store into LDS[buf] ----
#pragma unroll
      for (int i = 0; i < 4; ++i) {
        float4 v = ra[i];
        __bf16 h0, h1, h2, h3, l0, l1, l2, l3;
        split_hilo(v.x, h0, l0); split_hilo(v.y, h1, l1);
        split_hilo(v.z, h2, l2); split_hilo(v.w, h3, l3);
        uint2 hp, lp;
        hp.x = pack_bf16x2(h0, h1); hp.y = pack_bf16x2(h2, h3);
        lp.x = pack_bf16x2(l0, l1); lp.y = pack_bf16x2(l2, l3);
        *(uint2*)&AhiU[buf][a_row[i]][a_kc >> 1] = hp;
        *(uint2*)&AloU[buf][a_row[i]][a_kc >> 1] = lp;
      }
#pragma unroll
      for (int i = 0; i < 2; ++i) {
        float4 r0 = rb0[i], r1 = rb1[i];
        float c0[4] = {r0.x, r0.y, r0.z, r0.w};
        float c1[4] = {r1.x, r1.y, r1.z, r1.w};
#pragma unroll
        for (int j = 0; j < 4; ++j) {
          __bf16 h0, l0, h1, l1;
          split_hilo(c0[j], h0, l0);
          split_hilo(c1[j], h1, l1);
          BhiU[buf][b_nn + j][b_kk[i] >> 1] = pack_bf16x2(h0, h1); // {k,k+1}
          BloU[buf][b_nn + j][b_kk[i] >> 1] = pack_bf16x2(l0, l1);
        }
      }

      // ---- issue next K-tile global loads; latency overlaps barrier+WMMA ----
      if (ks + 1 < KSTEPS) {
        const int k0n = (ks + 1) * TILE_K;
#pragma unroll
        for (int i = 0; i < 4; ++i)
          ra[i] = *(const float4*)(X + (size_t)(m0 + a_row[i]) * IN_F + k0n + a_kc);
#pragma unroll
        for (int i = 0; i < 2; ++i) {
          const float* p0 = We + (size_t)(k0n + b_kk[i]) * OUT_F + n0 + b_nn;
          rb0[i] = *(const float4*)(p0);
          rb1[i] = *(const float4*)(p0 + OUT_F);
        }
      }

      wg_barrier_ds();   // DS-only wait: prefetch loads stay in flight

      // ---- A fragments (ISA 16-bit A layout: lane half -> K interleave of 8) ----
      FragU aHi[2], aLo[2];
      {
        const int d = (lane >> 4) << 2;           // dword 0 or 4 (K base 0 or 8)
#pragma unroll
        for (int mi = 0; mi < 2; ++mi) {
          const int m = wm * 32 + mi * 16 + (lane & 15);
          aHi[mi].q[0] = *(const uint4*)&AhiU[buf][m][d];       // K = kb..kb+7
          aHi[mi].q[1] = *(const uint4*)&AhiU[buf][m][8 + d];   // K = 16+kb..
          aLo[mi].q[0] = *(const uint4*)&AloU[buf][m][d];
          aLo[mi].q[1] = *(const uint4*)&AloU[buf][m][8 + d];
        }
      }
      // ---- B fragments + 3-term (hi*hi + lo*hi + hi*lo) WMMA accumulation ----
      {
        const int d16 = (lane >> 4) << 2;         // dword 0 or 4 (K base 0 or 8)
#pragma unroll
        for (int ni = 0; ni < 4; ++ni) {
          const int n = wn * 64 + ni * 16 + (lane & 15);
          FragU bHi, bLo;
          bHi.q[0] = *(const uint4*)&BhiU[buf][n][d16];
          bHi.q[1] = *(const uint4*)&BhiU[buf][n][d16 + 8];
          bLo.q[0] = *(const uint4*)&BloU[buf][n][d16];
          bLo.q[1] = *(const uint4*)&BloU[buf][n][d16 + 8];
#pragma unroll
          for (int mi = 0; mi < 2; ++mi) {
            acc[mi][ni] = __builtin_amdgcn_wmma_f32_16x16x32_bf16(
                false, aHi[mi].v, false, bHi.v, (short)0, acc[mi][ni], false, false);
            acc[mi][ni] = __builtin_amdgcn_wmma_f32_16x16x32_bf16(
                false, aLo[mi].v, false, bHi.v, (short)0, acc[mi][ni], false, false);
            acc[mi][ni] = __builtin_amdgcn_wmma_f32_16x16x32_bf16(
                false, aHi[mi].v, false, bLo.v, (short)0, acc[mi][ni], false, false);
          }
        }
      }
      // No second barrier: next iteration writes the other LDS buffer.
    }

    // ---- epilogue: bias add + segment-masked stores (C/D layout ISA 7.12.2) ----
#pragma unroll
    for (int ni = 0; ni < 4; ++ni) {
      const int ncol = n0 + wn * 64 + ni * 16 + (lane & 15);
      const float bv = Bias[e * OUT_F + ncol];
#pragma unroll
      for (int mi = 0; mi < 2; ++mi) {
        const int mbase = m0 + wm * 32 + mi * 16 + ((lane >> 4) << 3);
#pragma unroll
        for (int j = 0; j < 8; ++j) {
          const int row = mbase + j;
          if (row >= segLo && row < segHi)
            Y[(size_t)row * OUT_F + ncol] = acc[mi][ni][j] + bv;
        }
      }
    }
  }
}

extern "C" void kernel_launch(void* const* d_in, const int* in_sizes, int n_in,
                              void* d_out, int out_size, void* d_ws, size_t ws_size,
                              hipStream_t stream) {
  (void)in_sizes; (void)n_in; (void)out_size; (void)d_ws; (void)ws_size;
  const float* X    = (const float*)d_in[0];
  const int*   gsz  = (const int*)d_in[1];
  const float* W    = (const float*)d_in[2];
  const float* Bias = (const float*)d_in[3];
  float*       Y    = (float*)d_out;

  dim3 grid(OUT_F / TILE_N, T_TOKENS / TILE_M);  // (8, 64)
  dim3 block(256);                               // 8 wave32 waves
  hipLaunchKernelGGL(moe_grouped_gemm_bf16x3, grid, block, 0, stream,
                     X, gsz, W, Bias, Y);
}
